// LlamaAttention_cam_79551384256767
// MI455X (gfx1250) — matrix operand order repaired and checked by reading, not compiled, not run on hardware
//
#include <hip/hip_runtime.h>
#include <hip/hip_bf16.h>

// CDNA5 / gfx1250 implementation. wave32, WMMA f16 -> f32 accumulate.
// All GEMMs phrased as C = A * B^T with A:[M][K] f16 row-major, B:[N][K] f16
// row-major, so every fragment load is a contiguous 128-bit chunk matching the
// ISA 7.12.2 VGPR striping. Main GEMMs use 4x2 register blocking per wave
// (64x32 macro-tile, 8 v_wmma per 12 global_load_b128 -> ~21 FLOP/byte),
// sized to stay under the 256-VGPR direct-addressing budget (no spills).

typedef __attribute__((ext_vector_type(16))) _Float16 v16h;
typedef __attribute__((ext_vector_type(8)))  _Float16 v8h;
typedef __attribute__((ext_vector_type(8)))  float    v8f;

union AF { v16h v; v8h h[2]; };

#define T_     2048
#define DM_    2048
#define H_     16
#define D_     128
#define SB_    204      // int(0.1*2048)
#define EVICT_ 1229     // 2048 - int(0.4*2048)
#define RB_    819
#define LAST_  1792     // (T-1)//256*256
#define CLIP_  10000.0f

// ---------------------------------------------------------------------------
// Register-blocked GEMM: C = A * B^T, f32 out.
// Wave macro-tile: 64(M) x 32(N) = 4x2 WMMA tiles. Block: 8 waves as 2(M)x4(N)
// -> block tile 128 x 128. grid = (N/128, M/128). All call sites divide evenly.
// VGPR budget: 64 acc + 48 frags + addressing ~= 150 (< 256, no spills).
// A-frag (16x32): lane l -> row m0+(l&15); halves 0..7 = A[m,kb..kb+7],
//   halves 8..15 = A[m,kb+16..kb+23], kb=(l>>4)*8.
// B-frag (32x16 of row-major B[N][K]): lane l -> col n0+(l&15),
//   halves 0..7 = B[n,kb..kb+7], halves 8..15 = B[n,kb+16..kb+23], kb=(l>>4)*8.
// Waves whose whole 32-col span lies in [skip_lo, skip_hi) are skipped
// (fully-masked score columns -- softmax never reads them).
// ---------------------------------------------------------------------------
__global__ void __launch_bounds__(256, 1)
gemm_f16_f32_blk(const _Float16* __restrict__ A, int lda,
                 const _Float16* __restrict__ B, int ldb,
                 float* __restrict__ C, int ldc, int K,
                 float scale, int skip_lo, int skip_hi, int finalize) {
  int lane = threadIdx.x & 31;
  int wave = threadIdx.x >> 5;                 // 0..7
  int m0 = blockIdx.y * 128 + (wave & 1) * 64;
  int n0 = blockIdx.x * 128 + (wave >> 1) * 32;
  if (n0 >= skip_lo && n0 + 32 <= skip_hi) return;   // uniform per wave

  const _Float16* pa = A + (size_t)(m0 + (lane & 15)) * lda + (lane >> 4) * 8;
  const _Float16* pb = B + (size_t)(n0 + (lane & 15)) * ldb + (lane >> 4) * 8;
  const size_t arow = (size_t)16 * lda;
  const size_t brow = (size_t)16 * ldb;

  v8f acc[4][2] = {};
  for (int k = 0; k < K; k += 32) {
    __builtin_prefetch(pa + 256, 0, 3);        // global_prefetch_b8, 8 steps ahead
    __builtin_prefetch(pb + 256, 0, 3);
    AF a[4], b[2];
#pragma unroll
    for (int i = 0; i < 4; i++) {
      a[i].h[0] = *(const v8h*)(pa + i * arow);
      a[i].h[1] = *(const v8h*)(pa + i * arow + 16);
    }
#pragma unroll
    for (int j = 0; j < 2; j++) {
      b[j].h[0] = *(const v8h*)(pb + j * brow);
      b[j].h[1] = *(const v8h*)(pb + j * brow + 16);
    }
#pragma unroll
    for (int i = 0; i < 4; i++)
#pragma unroll
      for (int j = 0; j < 2; j++)
        acc[i][j] = __builtin_amdgcn_wmma_f32_16x16x32_f16(
            false, a[i].v, false, b[j].v, (short)0, acc[i][j], false, false);
    pa += 32; pb += 32;
  }

  int rbase = (lane >> 4) * 8;                 // C/D layout: VGPR r -> M=rbase+r
  int cbase = lane & 15;                       // N = lane%16
#pragma unroll
  for (int i = 0; i < 4; i++) {
#pragma unroll
    for (int j = 0; j < 2; j++) {
      int row0 = m0 + i * 16 + rbase;
      int col  = n0 + j * 16 + cbase;
#pragma unroll
      for (int r = 0; r < 8; r++) {
        float x = acc[i][j][r] * scale;
        if (finalize) {
          if (__builtin_isnan(x)) x = 0.0f;
          else if (__builtin_isinf(x)) x = x > 0.0f ? CLIP_ : -CLIP_;
        }
        C[(size_t)(row0 + r) * ldc + col] = x;
      }
    }
  }
}

// ---------------------------------------------------------------------------
// Simple per-tile k-loop (kept for the small P@V GEMM with its CAM epilogue).
// ---------------------------------------------------------------------------
__device__ __forceinline__ v8f wmma_loop(const _Float16* __restrict__ A, int lda,
                                         const _Float16* __restrict__ B, int ldb,
                                         int K, int lane) {
  const _Float16* pa = A + (size_t)(lane & 15) * lda + (lane >> 4) * 8;
  const _Float16* pb = B + (size_t)(lane & 15) * ldb + (lane >> 4) * 8;
  v8f acc = {};
  for (int k = 0; k < K; k += 32) {
    AF a, b;
    a.h[0] = *(const v8h*)(pa);
    a.h[1] = *(const v8h*)(pa + 16);
    b.h[0] = *(const v8h*)(pb);
    b.h[1] = *(const v8h*)(pb + 16);
    acc = __builtin_amdgcn_wmma_f32_16x16x32_f16(false, a.v, false, b.v,
                                                 (short)0, acc, false, false);
    pa += 32; pb += 32;
  }
  return acc;
}

// P @ V per head, f16 in / f32 acc, rank-1 CAM correction for q >= 1792,
// writes f16 into attn[t][h*128+d] for the final O-projection GEMM.
__global__ void pv_gemm(const _Float16* __restrict__ P,    // [T][T]
                        const _Float16* __restrict__ vT,   // [D][T] (this head)
                        const float* __restrict__ s_arr,   // [H][256]
                        const float* __restrict__ v_e,     // [H][D]
                        _Float16* __restrict__ attn, int h) {
  int lane = threadIdx.x & 31;
  int wave = threadIdx.x >> 5;
  int tile = blockIdx.x * 8 + wave;             // (T/16)*(D/16) = 1024 tiles
  int tm = tile >> 3, tn = tile & 7;            // Ntiles = 8
  v8f acc = wmma_loop(P + (size_t)tm * 16 * T_, T_,
                      vT + (size_t)tn * 16 * T_, T_, T_, lane);
  int row0 = tm * 16 + (lane >> 4) * 8;
  int col  = tn * 16 + (lane & 15);
  float ve = v_e[h * D_ + col];
#pragma unroll
  for (int r = 0; r < 8; r++) {
    int q = row0 + r;
    float x = acc[r];
    if (q >= LAST_) x += s_arr[h * 256 + (q - LAST_)] * ve;
    attn[(size_t)q * DM_ + h * D_ + col] = (_Float16)x;
  }
}

// ---------------------------------------------------------------------------
// Masked softmax over one score row; emits f16 probs (0 on masked columns)
// and, for q >= 1792, the tail mass s(q) = sum_{k>=1230} P[q,k].
// ---------------------------------------------------------------------------
__global__ void softmax_row(const float* __restrict__ S, _Float16* __restrict__ P,
                            float* __restrict__ s_arr, int h) {
  __shared__ float red[256];
  int row = blockIdx.x, tid = threadIdx.x;
  const float* srow = S + (size_t)row * T_;
  float m = -3.4e38f;
  for (int c = tid; c < T_; c += 256)
    if (c < SB_ || c >= EVICT_) m = fmaxf(m, srow[c]);
  red[tid] = m; __syncthreads();
  for (int s = 128; s > 0; s >>= 1) { if (tid < s) red[tid] = fmaxf(red[tid], red[tid + s]); __syncthreads(); }
  m = red[0]; __syncthreads();

  float sum = 0.0f, tail = 0.0f;
  for (int c = tid; c < T_; c += 256)
    if (c < SB_ || c >= EVICT_) {
      float e = expf(srow[c] - m);
      sum += e;
      if (c >= EVICT_ + 1) tail += e;
    }
  red[tid] = sum; __syncthreads();
  for (int s = 128; s > 0; s >>= 1) { if (tid < s) red[tid] += red[tid + s]; __syncthreads(); }
  sum = red[0]; __syncthreads();
  red[tid] = tail; __syncthreads();
  for (int s = 128; s > 0; s >>= 1) { if (tid < s) red[tid] += red[tid + s]; __syncthreads(); }
  tail = red[0]; __syncthreads();

  float inv = 1.0f / sum;
  _Float16* prow = P + (size_t)row * T_;
  for (int c = tid; c < T_; c += 256) {
    float p = (c < SB_ || c >= EVICT_) ? expf(srow[c] - m) * inv : 0.0f;
    prow[c] = (_Float16)p;
  }
  if (row >= LAST_ && tid == 0) s_arr[h * 256 + (row - LAST_)] = tail * inv;
}

// ---------------------------------------------------------------------------
// Exact JAX threefry-2x32 (20 rounds) for key(42), counter shape (16,).
// ---------------------------------------------------------------------------
__device__ __forceinline__ unsigned rotl32(unsigned x, int r) { return (x << r) | (x >> (32 - r)); }
__device__ void threefry2x32(unsigned k0, unsigned k1, unsigned c0, unsigned c1,
                             unsigned* o0, unsigned* o1) {
  unsigned ks2 = 0x1BD11BDAu ^ k0 ^ k1;
  unsigned x0 = c0 + k0, x1 = c1 + k1;
  const int r0[4] = {13, 15, 26, 6}, r1[4] = {17, 29, 16, 24};
#pragma unroll
  for (int j = 0; j < 4; j++) { x0 += x1; x1 = rotl32(x1, r0[j]); x1 ^= x0; }
  x0 += k1; x1 += ks2 + 1u;
#pragma unroll
  for (int j = 0; j < 4; j++) { x0 += x1; x1 = rotl32(x1, r1[j]); x1 ^= x0; }
  x0 += ks2; x1 += k0 + 2u;
#pragma unroll
  for (int j = 0; j < 4; j++) { x0 += x1; x1 = rotl32(x1, r0[j]); x1 ^= x0; }
  x0 += k0; x1 += k1 + 3u;
#pragma unroll
  for (int j = 0; j < 4; j++) { x0 += x1; x1 = rotl32(x1, r1[j]); x1 ^= x0; }
  x0 += k1; x1 += ks2 + 4u;
#pragma unroll
  for (int j = 0; j < 4; j++) { x0 += x1; x1 = rotl32(x1, r0[j]); x1 ^= x0; }
  x0 += ks2; x1 += k0 + 5u;
  *o0 = x0; *o1 = x1;
}

// CAM gate for one head: mean tail attention of last query row, Bernoulli draw,
// v_e = v[1229,:] * bern / rb.
__global__ void cam_kernel(const _Float16* __restrict__ P,     // [T][T]
                           const float* __restrict__ v1229,    // [H][D]
                           float* __restrict__ v_e, int h) {
  __shared__ float red[256];
  __shared__ float sh_bern;
  int tid = threadIdx.x;
  const _Float16* last = P + (size_t)(T_ - 1) * T_;
  float s = 0.0f;
  for (int c = EVICT_ + 1 + tid; c < T_; c += 256) s += (float)last[c];
  red[tid] = s; __syncthreads();
  for (int st = 128; st > 0; st >>= 1) { if (tid < st) red[tid] += red[tid + st]; __syncthreads(); }
  if (tid == 0) {
    float mean = red[0] / 818.0f + 1e-6f;
    float p = (float)last[EVICT_] / mean;
    if (__builtin_isnan(p)) p = 0.0f;
    p = fminf(fmaxf(p, 0.0f), 1.0f);
    unsigned c0 = (unsigned)(h & 7), c1 = c0 + 8u;   // iota(16) split halves
    unsigned o0, o1; threefry2x32(0u, 42u, c0, c1, &o0, &o1);
    unsigned bits = (h < 8) ? o0 : o1;
    float u = __uint_as_float((bits >> 9) | 0x3f800000u) - 1.0f;
    sh_bern = (u < p) ? 1.0f : 0.0f;
  }
  __syncthreads();
  if (tid < D_) v_e[h * D_ + tid] = v1229[h * D_ + tid] * sh_bern / (float)RB_;
}

// ---------------------------------------------------------------------------
// RoPE + head-layout: raw[T][Dm] f32 -> dst16[h][t][d] f16
// ---------------------------------------------------------------------------
__global__ void rope_kernel(const float* __restrict__ raw, _Float16* __restrict__ dst) {
  int t = blockIdx.x, h = blockIdx.y, d = threadIdx.x;
  float x     = raw[(size_t)t * DM_ + h * D_ + d];
  float other = raw[(size_t)t * DM_ + h * D_ + (d < 64 ? d + 64 : d - 64)];
  float rot = (d < 64) ? -other : other;
  float invf = __powf(10000.0f, -((float)(d & 63)) / 64.0f);
  float c, s;
  sincosf((float)t * invf, &s, &c);
  dst[((size_t)h * T_ + t) * D_ + d] = (_Float16)(x * c + rot * s);
}

// v: raw[T][Dm] f32 -> vT[h][d][t] f16 (transposed per head); also grab row 1229.
__global__ void v_layout(const float* __restrict__ raw, _Float16* __restrict__ vT,
                         float* __restrict__ v1229) {
  int t = blockIdx.x, h = blockIdx.y, d = threadIdx.x;
  float v = raw[(size_t)t * DM_ + h * D_ + d];
  vT[((size_t)h * D_ + d) * T_ + t] = (_Float16)v;
  if (t == EVICT_) v1229[h * D_ + d] = v;
}

// f32 -> f16, optional nan_to_num (for hidden_states).
__global__ void f32_to_f16_k(const float* __restrict__ x, _Float16* __restrict__ y,
                             int n, int sanitize) {
  int i = blockIdx.x * 256 + threadIdx.x;
  if (i >= n) return;
  float v = x[i];
  if (sanitize) {
    if (__builtin_isnan(v)) v = 0.0f;
    else if (__builtin_isinf(v)) v = v > 0.0f ? CLIP_ : -CLIP_;
  }
  y[i] = (_Float16)v;
}

// ---------------------------------------------------------------------------
extern "C" void kernel_launch(void* const* d_in, const int* in_sizes, int n_in,
                              void* d_out, int out_size, void* d_ws, size_t ws_size,
                              hipStream_t stream) {
  const float* hs = (const float*)d_in[0];
  const float* Wq = (const float*)d_in[1];
  const float* Wk = (const float*)d_in[2];
  const float* Wv = (const float*)d_in[3];
  const float* Wo = (const float*)d_in[4];
  float* out = (float*)d_out;

  const size_t NE = (size_t)T_ * DM_;           // 4M elements
  char* p = (char*)d_ws;
  auto carve = [&](size_t bytes) { void* r = (void*)p; p += (bytes + 255) & ~(size_t)255; return r; };

  _Float16* hs16   = (_Float16*)carve(NE * 2);
  _Float16* wq16   = (_Float16*)carve(NE * 2);
  _Float16* wk16   = (_Float16*)carve(NE * 2);
  _Float16* wv16   = (_Float16*)carve(NE * 2);
  _Float16* wo16   = (_Float16*)carve(NE * 2);
  float*    craw   = (float*)carve(NE * 4);     // reused: q_raw -> k_raw -> v_raw -> scores
  _Float16* q16    = (_Float16*)carve(NE * 2);  // [H][T][D]
  _Float16* k16    = (_Float16*)carve(NE * 2);  // [H][T][D]
  _Float16* vT16   = (_Float16*)carve(NE * 2);  // [H][D][T]
  _Float16* P16    = (_Float16*)carve(NE * 2);  // per-head probs [T][T]
  _Float16* attn16 = (_Float16*)carve(NE * 2);  // [T][H*D]
  float*    v1229  = (float*)carve(H_ * D_ * 4);
  float*    v_e    = (float*)carve(H_ * D_ * 4);
  float*    s_arr  = (float*)carve(H_ * 256 * 4);

  int cgrid = (int)(NE / 256);
  f32_to_f16_k<<<cgrid, 256, 0, stream>>>(hs, hs16, (int)NE, 1);
  f32_to_f16_k<<<cgrid, 256, 0, stream>>>(Wq, wq16, (int)NE, 0);
  f32_to_f16_k<<<cgrid, 256, 0, stream>>>(Wk, wk16, (int)NE, 0);
  f32_to_f16_k<<<cgrid, 256, 0, stream>>>(Wv, wv16, (int)NE, 0);
  f32_to_f16_k<<<cgrid, 256, 0, stream>>>(Wo, wo16, (int)NE, 0);

  const dim3 gfull(DM_ / 128, T_ / 128);        // (16, 16) for 2048x2048 output
  // Q projection + RoPE
  gemm_f16_f32_blk<<<gfull, 256, 0, stream>>>(hs16, DM_, wq16, DM_, craw, DM_,
                                              DM_, 1.0f, 0, 0, 0);
  rope_kernel<<<dim3(T_, H_), D_, 0, stream>>>(craw, q16);
  // K projection + RoPE
  gemm_f16_f32_blk<<<gfull, 256, 0, stream>>>(hs16, DM_, wk16, DM_, craw, DM_,
                                              DM_, 1.0f, 0, 0, 0);
  rope_kernel<<<dim3(T_, H_), D_, 0, stream>>>(craw, k16);
  // V projection + transpose layout
  gemm_f16_f32_blk<<<gfull, 256, 0, stream>>>(hs16, DM_, wv16, DM_, craw, DM_,
                                              DM_, 1.0f, 0, 0, 0);
  v_layout<<<dim3(T_, H_), D_, 0, stream>>>(craw, vT16, v1229);

  float* S = craw;                              // scores scratch (per head)
  const float scale = 0.08838834764831845f;     // 1/sqrt(128)
  const int ptiles = (T_ / 16) * (D_ / 16);     // 1024
  for (int h = 0; h < H_; h++) {
    const _Float16* qh = q16 + (size_t)h * T_ * D_;
    const _Float16* kh = k16 + (size_t)h * T_ * D_;
    // S = Q K^T * scale, skipping waves fully inside masked columns [204,1229)
    gemm_f16_f32_blk<<<dim3(T_ / 128, T_ / 128), 256, 0, stream>>>(
        qh, D_, kh, D_, S, T_, D_, scale, SB_, EVICT_, 0);
    softmax_row<<<T_, 256, 0, stream>>>(S, P16, s_arr, h);
    cam_kernel<<<1, 256, 0, stream>>>(P16, v1229, v_e, h);
    pv_gemm<<<ptiles / 8, 256, 0, stream>>>(P16, vT16 + (size_t)h * D_ * T_,
                                            s_arr, v_e, attn16, h);
  }

  // out = attn @ Wo^T, nan_to_num clip
  gemm_f16_f32_blk<<<gfull, 256, 0, stream>>>(attn16, DM_, wo16, DM_, out, DM_,
                                              DM_, 1.0f, 0, 0, 1);
}
